// PhonemeGRU_24043226923218
// MI455X (gfx1250) — matrix operand order repaired and checked by reading, not compiled, not run on hardware
//
#include <hip/hip_runtime.h>

typedef __attribute__((ext_vector_type(16))) _Float16 v16h;
typedef __attribute__((ext_vector_type(8)))  float    v8f;

#define VOCAB 45
#define E     16
#define H     32
#define G3    96      // 3*H
#define SEQ   512
#define OUTD  64

#define LOG2E 1.44269504088896340736f

__device__ __forceinline__ float fast_sigmoid_pre(float x_log2e) {
    // x already scaled by log2(e): sigmoid = 1 / (1 + 2^(-x*log2e))
    return __builtin_amdgcn_rcpf(1.0f + __builtin_amdgcn_exp2f(-x_log2e));
}
__device__ __forceinline__ float fast_tanh_pre(float x_2log2e) {
    // x already scaled by 2*log2(e): tanh = 1 - 2 / (1 + 2^(2y*log2e))
    return 1.0f - 2.0f * __builtin_amdgcn_rcpf(1.0f + __builtin_amdgcn_exp2f(x_2log2e));
}

__global__ __launch_bounds__(128) void phoneme_gru_kernel(
    const int*   __restrict__ x,      // (B, S) int32
    const float* __restrict__ emb,    // (VOCAB, E)
    const float* __restrict__ w_ih,   // (3H, E)
    const float* __restrict__ w_hh,   // (3H, H)
    const float* __restrict__ b_ih,   // (3H)
    const float* __restrict__ b_hh,   // (3H)
    const float* __restrict__ fc_w,   // (O, H)
    const float* __restrict__ fc_b,   // (O)
    float*       __restrict__ out)    // (B, O)
{
    __shared__ __align__(16) float s_gi[VOCAB * G3];   // 17,280 B fused input-gate LUT

    const int tid  = threadIdx.x;
    const int lane = tid & 31;
    const int wave = tid >> 5;
    const int hi   = lane >> 4;   // lane half (0/1)
    const int nl   = lane & 15;   // N index / batch-in-tile / M-in-tile for A builds

    // ---- Fused input projection table, pre-scaled for exp2-based activations:
    //   r,z rows (g < 64): (emb·w_ih + b_ih + b_hh) * log2e
    //   n   rows (g >= 64): (emb·w_ih + b_ih) * 2*log2e   (b_hh kept in WMMA C operand)
    for (int idx = tid; idx < VOCAB * G3; idx += blockDim.x) {
        int v = idx / G3, g = idx - v * G3;
        float acc = b_ih[g] + (g < 2 * H ? b_hh[g] : 0.0f);
        #pragma unroll
        for (int e = 0; e < E; ++e) acc += emb[v * E + e] * w_ih[g * E + e];
        s_gi[idx] = acc * (g < 2 * H ? LOG2E : 2.0f * LOG2E);
    }
    __syncthreads();

    // ---- Static A fragments: w_hh as six 16(gate) x 32(hidden) f16 tiles, pre-scaled.
    // A 16x32 f16 layout: lane: M = lane%16; halves i<8 -> K = hi*8+i, i>=8 -> K = 16+hi*8+(i-8).
    v16h a_hh[6];
    #pragma unroll
    for (int mt = 0; mt < 6; ++mt) {
        const float scl = (mt < 4) ? LOG2E : 2.0f * LOG2E;
        const float* wrow = w_hh + (mt * 16 + nl) * H;
        #pragma unroll
        for (int i = 0; i < 8; ++i) {
            a_hh[mt][i]     = (_Float16)(scl * wrow[hi * 8 + i]);
            a_hh[mt][i + 8] = (_Float16)(scl * wrow[16 + hi * 8 + i]);
        }
    }
    // C bias fragments for the two n-gate tiles (mt = 4,5), pre-scaled by 2*log2e.
    v8f cb0 = {}, cb1 = {};
    #pragma unroll
    for (int v = 0; v < 8; ++v) {
        cb0[v] = 2.0f * LOG2E * b_hh[4 * 16 + hi * 8 + v];
        cb1[v] = 2.0f * LOG2E * b_hh[5 * 16 + hi * 8 + v];
    }

    const int  base = blockIdx.x * 64 + wave * 16;     // 16 batch rows per wave
    const int* xrow = x + (base + nl) * SEQ;           // this lane's token stream (batch = nl)

    v8f h0 = {}, h1 = {};   // hidden state, j-tiles 0 (j=0..15) and 1 (j=16..31), D layout

    #pragma unroll 1
    for (int t = 0; t < SEQ; ++t) {
        const int tok = xrow[t];

        // gi gather: 8 consecutive floats per gate-tile, float4-vectorized LDS loads
        const float4* grow = (const float4*)(s_gi + tok * G3 + hi * 8);
        float gi[6][8];
        #pragma unroll
        for (int mt = 0; mt < 6; ++mt) {
            float4 g0 = grow[mt * 4];
            float4 g1 = grow[mt * 4 + 1];
            gi[mt][0] = g0.x; gi[mt][1] = g0.y; gi[mt][2] = g0.z; gi[mt][3] = g0.w;
            gi[mt][4] = g1.x; gi[mt][5] = g1.y; gi[mt][6] = g1.z; gi[mt][7] = g1.w;
        }

        // Build B = h^T fragment (32(K=j) x 16(N=batch) f16) directly from D-layout h (the p0/p1 shuffles go unused).
        // B lane: N = lane%16, halves i<8 -> K = hi*8 + i, i>=8 -> K = 16 + hi*8 + (i-8).
        v8f p0, p1;
        #pragma unroll
        for (int v = 0; v < 8; ++v) {
            p0[v] = __shfl_xor(h0[v], 16, 32);
            p1[v] = __shfl_xor(h1[v], 16, 32);
        }
        v16h bh;
        #pragma unroll
        for (int i = 0; i < 8; ++i) {
            bh[i]     = (_Float16)h0[i];   // K = hi*8 + i
            bh[i + 8] = (_Float16)h1[i];   // K = 16 + hi*8 + i
        }

        // gh^T = w_hh x h^T : six independent 16x16x32 f16 WMMAs (f32 accum)
        v8f zc = {};
        v8f d0 = __builtin_amdgcn_wmma_f32_16x16x32_f16(false, a_hh[0], false, bh, (short)0, zc,  false, false);
        v8f d1 = __builtin_amdgcn_wmma_f32_16x16x32_f16(false, a_hh[1], false, bh, (short)0, zc,  false, false);
        v8f d2 = __builtin_amdgcn_wmma_f32_16x16x32_f16(false, a_hh[2], false, bh, (short)0, zc,  false, false);
        v8f d3 = __builtin_amdgcn_wmma_f32_16x16x32_f16(false, a_hh[3], false, bh, (short)0, zc,  false, false);
        v8f d4 = __builtin_amdgcn_wmma_f32_16x16x32_f16(false, a_hh[4], false, bh, (short)0, cb0, false, false);
        v8f d5 = __builtin_amdgcn_wmma_f32_16x16x32_f16(false, a_hh[5], false, bh, (short)0, cb1, false, false);

        // Gate elementwise — branch-free, exp2/rcp hardware transcendentals,
        // log2e factors already folded into gi/A/C.
        #pragma unroll
        for (int v = 0; v < 8; ++v) {
            // j-tile 0
            float r = fast_sigmoid_pre(gi[0][v] + d0[v]);
            float z = fast_sigmoid_pre(gi[2][v] + d2[v]);
            float n = fast_tanh_pre(gi[4][v] + r * d4[v]);
            h0[v] = n + z * (h0[v] - n);
            // j-tile 1
            r = fast_sigmoid_pre(gi[1][v] + d1[v]);
            z = fast_sigmoid_pre(gi[3][v] + d3[v]);
            n = fast_tanh_pre(gi[5][v] + r * d5[v]);
            h1[v] = n + z * (h1[v] - n);
        }
    }

    // ---- Final FC: out^T = fc_w x h^T (+ fc_b), four more WMMAs (unscaled weights)
    v8f p0, p1;
    #pragma unroll
    for (int v = 0; v < 8; ++v) {
        p0[v] = __shfl_xor(h0[v], 16, 32);
        p1[v] = __shfl_xor(h1[v], 16, 32);
    }
    v16h bh;
    #pragma unroll
    for (int i = 0; i < 8; ++i) {
        bh[i]     = (_Float16)h0[i];
        bh[i + 8] = (_Float16)h1[i];
    }

    #pragma unroll
    for (int mt = 0; mt < 4; ++mt) {
        v16h afc;
        const float* frow = fc_w + (mt * 16 + nl) * H;
        #pragma unroll
        for (int i = 0; i < 8; ++i) {
            afc[i]     = (_Float16)frow[hi * 8 + i];
            afc[i + 8] = (_Float16)frow[16 + hi * 8 + i];
        }
        v8f cfc;
        #pragma unroll
        for (int v = 0; v < 8; ++v) cfc[v] = fc_b[mt * 16 + hi * 8 + v];

        v8f d = __builtin_amdgcn_wmma_f32_16x16x32_f16(false, afc, false, bh, (short)0, cfc, false, false);

        #pragma unroll
        for (int v = 0; v < 8; ++v)
            out[(base + nl) * OUTD + mt * 16 + hi * 8 + v] = d[v];
    }
}

extern "C" void kernel_launch(void* const* d_in, const int* in_sizes, int n_in,
                              void* d_out, int out_size, void* d_ws, size_t ws_size,
                              hipStream_t stream) {
    const int*   x    = (const int*)  d_in[0];
    const float* emb  = (const float*)d_in[1];
    const float* w_ih = (const float*)d_in[2];
    const float* w_hh = (const float*)d_in[3];
    const float* b_ih = (const float*)d_in[4];
    const float* b_hh = (const float*)d_in[5];
    const float* fc_w = (const float*)d_in[6];
    const float* fc_b = (const float*)d_in[7];
    float* out = (float*)d_out;

    const int B = in_sizes[0] / SEQ;            // 8192
    const int blocks = B / 64;                  // 4 waves/block, 16 rows/wave
    phoneme_gru_kernel<<<blocks, 128, 0, stream>>>(
        x, emb, w_ih, w_hh, b_ih, b_hh, fc_w, fc_b, out);
}
